// ElmanLeakySelective_7206955123459
// MI455X (gfx1250) — matrix operand order repaired and checked by reading, not compiled, not run on hardware
//
#include <hip/hip_runtime.h>
#include <hip/hip_bf16.h>

// Problem sizes (fixed by the reference)
#define TT 2048
#define BB 16
#define DD 1024
#define MM (TT * BB)   // 32768 rows for the big GEMMs

typedef __attribute__((ext_vector_type(16))) __bf16     v16bf;
typedef __attribute__((ext_vector_type(8)))  float      v8f;
typedef __attribute__((ext_vector_type(4)))  unsigned   u32x4;
typedef __attribute__((ext_vector_type(4)))  int        i32x4;
typedef __attribute__((ext_vector_type(8)))  int        i32x8;

// ---------- bf16 helpers (bit-level; avoids __bf16 arithmetic) -------------
__device__ __forceinline__ unsigned short f2bf(float f) {
  unsigned u = __float_as_uint(f);
  u += 0x7fffu + ((u >> 16) & 1u);          // round-to-nearest-even
  return (unsigned short)(u >> 16);
}
__device__ __forceinline__ float bf2f(unsigned short h) {
  return __uint_as_float(((unsigned)h) << 16);
}

union FragU { u32x4 q[2]; v16bf v; };

// A-matrix 16x32 bf16 fragment (ISA 7.12.2): lane m = lane&15, half = lane>>4
//   halfwords 0-7  -> K = k0 + 8*half + (0..7)
//   halfwords 8-15 -> K = k0 + 16 + 8*half + (0..7)
__device__ __forceinline__ v16bf frag_a(const unsigned short* p, int stride,
                                        int k0, int lane) {
  const unsigned short* rp = p + (size_t)(lane & 15) * stride + k0 + ((lane >> 4) << 3);
  FragU u;
  u.q[0] = *(const u32x4*)rp;
  u.q[1] = *(const u32x4*)(rp + 16);
  return u.v;
}

// B-matrix 32x16 bf16 fragment: lane n = lane&15, half = lane>>4
//   halfwords 0-7 -> K = k0 + 8*half + i; halfwords 8-15 -> K = k0 + 16 + 8*half + (i-8)
// p points at [n][k] rows (row-major weight W[e,d], e = output col, d = K)
__device__ __forceinline__ v16bf frag_b(const unsigned short* p, int stride,
                                        int k0, int lane) {
  const unsigned short* rp = p + (size_t)(lane & 15) * stride + k0 + ((lane >> 4) << 3);
  FragU u;
  u.q[0] = *(const u32x4*)rp;
  u.q[1] = *(const u32x4*)(rp + 16);
  return u.v;
}

// ---------- Tensor Data Mover: 2D bf16 tile, global -> LDS -----------------
// Builds the Tensor DMA Descriptor (D#) per ISA 08 §8.3/§8.4 and issues
// TENSOR_LOAD_TO_LDS (VIMAGE op 196) with 2 SGPR groups (<=2D tensor,
// VADDR2/VADDR3 NULL). Tracked by TENSORcnt.
//   group0: [1:0]=count=1, [63:32]=lds byte addr, [120:64]=global byte addr,
//           [127:126]=type=2
//   group1: [17:16]=data_size=1 (2B), tensor_dim0/1, tile_dim0/1,
//           tensor_dim0_stride; workgroup_mask=0 (not clustered)
__device__ __forceinline__ void tdm_load_2d(unsigned lds_byte_off,
                                            const void* gptr,
                                            unsigned rows, unsigned cols,
                                            unsigned stride_elems) {
  unsigned long long ga = (unsigned long long)gptr;
  i32x4 g0;
  g0[0] = 1;                                              // count=1, user mode
  g0[1] = (int)lds_byte_off;                              // lds_addr
  g0[2] = (int)(unsigned)(ga & 0xffffffffu);              // global_addr[31:0]
  g0[3] = (int)((unsigned)((ga >> 32) & 0x01ffffffu) | (2u << 30)); // +type=2
  i32x8 g1;
  g1[0] = (int)(1u << 16);                                // data_size=1 (2B)
  g1[1] = (int)((cols & 0xffffu) << 16);                  // tensor_dim0[15:0]
  g1[2] = (int)(((cols >> 16) & 0xffffu) | ((rows & 0xffffu) << 16)); // dim0 hi | dim1 lo
  g1[3] = (int)(((rows >> 16) & 0xffffu) | ((cols & 0xffffu) << 16)); // dim1 hi | tile_dim0
  g1[4] = (int)(rows & 0xffffu);                          // tile_dim1 (tile_dim2=0)
  g1[5] = (int)stride_elems;                              // tensor_dim0_stride lo32
  g1[6] = 0;
  g1[7] = 0;
  asm volatile("tensor_load_to_lds %0, %1" :: "s"(g0), "s"(g1) : "memory");
}

// ---------- elementwise kernels -------------------------------------------
__global__ void cvt_f32_bf16_kernel(const float* __restrict__ src,
                                    unsigned short* __restrict__ dst, size_t n) {
  size_t i = (size_t)blockIdx.x * blockDim.x + threadIdx.x;
  size_t st = (size_t)gridDim.x * blockDim.x;
  for (; i < n; i += st) dst[i] = f2bf(src[i]);
}

__global__ void decay_kernel(const float* __restrict__ A,
                             float* __restrict__ decay) {
  int i = threadIdx.x + blockIdx.x * blockDim.x;
  if (i < DD) decay[i] = __expf(-__expf(A[i]));
}

// init: zero barrier state, seed both h ping-pong buffers with h0 (bf16)
__global__ void init_kernel(const float* __restrict__ h0,
                            unsigned short* __restrict__ hbuf,
                            unsigned* __restrict__ bar) {
  int i = threadIdx.x + blockIdx.x * blockDim.x;
  if (i < 2) bar[i] = 0u;
  if (i < BB * DD) {
    unsigned short v = f2bf(h0[i]);
    hbuf[i] = v;
    hbuf[BB * DD + i] = v;
  }
}

// ---------- fused GEMM: C = A(bf16) @ W(bf16)^T with mode-specific epilogue
// mode 0: outf = acc + bias[col]                                  (wx)
// mode 1: outf = exp(-softplus(acc + bias[col]) * decay[col])     (alpha)
// mode 2: outb = bf16(acc + bias[col])                            (gx + b_gate)
// mode 3: g = acc + bf2f(gx[idx]); outf = bf2f(h[idx]) * silu(g)  (final gate)
#define GMT 128
#define GNT 128
#define GKT 32
__global__ __launch_bounds__(256) void gemm_epi_kernel(
    const unsigned short* __restrict__ Abf,   // [MM, DD] bf16
    const unsigned short* __restrict__ Wbf,   // [DD, DD] bf16  (row = out col)
    const float* __restrict__ bias,           // [DD] or nullptr
    const float* __restrict__ decay,          // mode 1
    const unsigned short* __restrict__ gx,    // mode 3, bf16
    const unsigned short* __restrict__ hbt,   // mode 3, bf16
    float* __restrict__ outf,
    unsigned short* __restrict__ outb,
    int mode) {
  __shared__ unsigned short sA[GMT * GKT];    // 8 KB
  __shared__ unsigned short sB[GNT * GKT];    // 8 KB

  const int tid = threadIdx.x;
  const int lane = tid & 31, wid = tid >> 5;
  const int bm = blockIdx.x >> 3;             // 256 m-blocks
  const int bn = blockIdx.x & 7;              // 8 n-blocks
  const int m0 = bm * GMT, n0 = bn * GNT;
  const int wm = (wid & 3) * 32;              // wave 32-row strip
  const int wn = (wid >> 2) * 64;             // wave 64-col strip

  v8f acc[2][4] = {};

  const int rowL = tid >> 1;                  // 0..127
  const int segL = (tid & 1) * 16;            // 0 or 16 halfwords

  for (int kk = 0; kk < DD; kk += GKT) {
    *(i32x8*)&sA[rowL * GKT + segL] =
        *(const i32x8*)&Abf[(size_t)(m0 + rowL) * DD + kk + segL];
    *(i32x8*)&sB[rowL * GKT + segL] =
        *(const i32x8*)&Wbf[(size_t)(n0 + rowL) * DD + kk + segL];
    __syncthreads();

    v16bf a[2], b[4];
#pragma unroll
    for (int i = 0; i < 2; ++i)
      a[i] = frag_a(&sA[(wm + i * 16) * GKT], GKT, 0, lane);
#pragma unroll
    for (int j = 0; j < 4; ++j)
      b[j] = frag_b(&sB[(wn + j * 16) * GKT], GKT, 0, lane);
#pragma unroll
    for (int i = 0; i < 2; ++i)
#pragma unroll
      for (int j = 0; j < 4; ++j)
        acc[i][j] = __builtin_amdgcn_wmma_f32_16x16x32_bf16(
            false, a[i], false, b[j], (short)0, acc[i][j], false, false);
    __syncthreads();
  }

  const int half = lane >> 4;
#pragma unroll
  for (int i = 0; i < 2; ++i) {
#pragma unroll
    for (int j = 0; j < 4; ++j) {
      const int col = n0 + wn + j * 16 + (lane & 15);
#pragma unroll
      for (int r = 0; r < 8; ++r) {
        const int row = m0 + wm + i * 16 + r + 8 * half;
        const size_t idx = (size_t)row * DD + col;
        float t = acc[i][j][r];
        if (bias) t += bias[col];
        if (mode == 0) {
          outf[idx] = t;
        } else if (mode == 1) {
          float sp = (t > 20.f) ? t : log1pf(__expf(t));
          outf[idx] = __expf(-sp * decay[col]);
        } else if (mode == 2) {
          outb[idx] = f2bf(t);
        } else {
          float g = t + bf2f(gx[idx]);
          float sig = 1.f / (1.f + __expf(-g));
          outf[idx] = bf2f(hbt[idx]) * (g * sig);
        }
      }
    }
  }
}

// ---------- persistent sequential scan ------------------------------------
// 16 WGs x 256 threads (8 waves). WG wg owns columns [wg*64, wg*64+64):
//   LDS: R slice [64][1024] bf16 (128KB, TDM-staged once, resident)
//        h stage [16][1024] bf16 (32KB, TDM-restaged per step)
//        partials 4*32*8 f32 (4KB)
// per step: 4 col-tiles x (2 waves split K) -> 16 chained WMMAs/wave,
// LDS K-reduction, tanh/blend epilogue, bf16 h write, device barrier,
// then TENSOR_LOAD_TO_LDS pulls the new h tile (tracked by TENSORcnt).
#define SCAN_WGS 16
#define SH_OFF (64 * 1024 * 2)                // byte offset of sH in LDS
#define SCAN_SMEM (64 * 1024 * 2 + 16 * 1024 * 2 + 4 * 32 * 8 * 4)

__global__ __launch_bounds__(256) void scan_kernel(
    const unsigned short* __restrict__ Rbf,   // [DD, DD] bf16
    const float* __restrict__ wx,             // [MM, DD]
    const float* __restrict__ alpha,          // [MM, DD]
    unsigned short* __restrict__ hbt,         // out trace [MM, DD] bf16
    unsigned short* __restrict__ hbuf,        // [2][BB*DD] bf16 ping-pong
    unsigned* __restrict__ bar) {             // {counter, generation}
  extern __shared__ unsigned short smem[];
  unsigned short* sR = smem;                        // 64*1024 halfwords
  unsigned short* sH = smem + 64 * 1024;            // 16*1024 halfwords
  float* sP = (float*)(smem + 64 * 1024 + 16 * 1024);

  const int tid = threadIdx.x, lane = tid & 31, wid = tid >> 5;
  const int wg = blockIdx.x;
  const int c0 = wg * 64;
  const int ns = wid >> 1;                    // col tile 0..3
  const int kh = wid & 1;                     // K half 0..1
  const int half = lane >> 4;

  // TDM-stage this WG's R slice (64x1024, resident) and the initial h tile.
  if (wid == 0) {
    tdm_load_2d(0, Rbf + (size_t)c0 * DD, 64, DD, DD);
    tdm_load_2d(SH_OFF, hbuf, BB, DD, DD);
    __builtin_amdgcn_s_wait_tensorcnt(0);
  }
  __syncthreads();

  for (int t = 0; t < TT; ++t) {
    // cand pre-activation partial: h @ R^T over this wave's K half
    v8f acc = {};
    const int kbase = kh * 512;
#pragma unroll 4
    for (int kc = 0; kc < 16; ++kc) {
      v16bf a = frag_a(sH, DD, kbase + kc * 32, lane);
      v16bf b = frag_b(&sR[(ns * 16) * DD], DD, kbase + kc * 32, lane);
      acc = __builtin_amdgcn_wmma_f32_16x16x32_bf16(
          false, a, false, b, (short)0, acc, false, false);
    }
    if (kh == 1) {
      float* dst = &sP[(ns * 32 + lane) * 8];
#pragma unroll
      for (int r = 0; r < 8; ++r) dst[r] = acc[r];
    }
    __syncthreads();

    if (kh == 0) {
      const float* src = &sP[(ns * 32 + lane) * 8];
      const int colg = c0 + ns * 16 + (lane & 15);
#pragma unroll
      for (int r = 0; r < 8; ++r) {
        const int brow = r + 8 * half;
        const size_t idx = ((size_t)t * BB + brow) * DD + colg;
        float cand = tanhf(acc[r] + src[r] + wx[idx]);
        float al = alpha[idx];
        float hold = bf2f(sH[brow * DD + colg]);
        float hn = al * hold + (1.f - al) * cand;
        unsigned short hb = f2bf(hn);
        hbt[idx] = hb;
        hbuf[(size_t)((t + 1) & 1) * (BB * DD) + brow * DD + colg] = hb;
      }
      // warm next step's operands into GL2 (global_prefetch_b8)
      if (lane == 0 && t + 1 < TT) {
        __builtin_prefetch(&wx[((size_t)(t + 1) * BB) * DD + c0], 0, 0);
        __builtin_prefetch(&alpha[((size_t)(t + 1) * BB) * DD + c0], 0, 0);
      }
    }

    __threadfence();          // release h writes to device scope
    __syncthreads();
    // Cluster barrier: real sync if dispatched as a 16-WG cluster,
    // architecturally S_NOP otherwise (ISA 08 §2).
    __builtin_amdgcn_s_cluster_barrier();
    // Portable device-wide ticket barrier across the 16 persistent WGs.
    if (tid == 0) {
      unsigned ticket = atomicAdd(&bar[0], 1u) + 1u;
      unsigned target = (ticket + (SCAN_WGS - 1u)) / SCAN_WGS;
      if ((ticket % SCAN_WGS) == 0u) atomicAdd(&bar[1], 1u);
      while (__hip_atomic_load(&bar[1], __ATOMIC_ACQUIRE,
                               __HIP_MEMORY_SCOPE_AGENT) < target) {
        __builtin_amdgcn_s_sleep(1);
      }
    }
    __syncthreads();

    // TDM-restage h for step t+1 from the buffer everyone just wrote.
    if (wid == 0) {
      tdm_load_2d(SH_OFF, hbuf + (size_t)((t + 1) & 1) * (BB * DD), BB, DD, DD);
      __builtin_amdgcn_s_wait_tensorcnt(0);
    }
    __syncthreads();
  }
}

// ---------- host orchestration --------------------------------------------
extern "C" void kernel_launch(void* const* d_in, const int* in_sizes, int n_in,
                              void* d_out, int out_size, void* d_ws,
                              size_t ws_size, hipStream_t stream) {
  const float* x     = (const float*)d_in[0];
  const float* h0    = (const float*)d_in[1];
  const float* Wx    = (const float*)d_in[2];
  const float* R     = (const float*)d_in[3];
  const float* bias  = (const float*)d_in[4];
  const float* Wd    = (const float*)d_in[5];
  const float* bd    = (const float*)d_in[6];
  const float* Avec  = (const float*)d_in[7];
  const float* Wgx   = (const float*)d_in[8];
  const float* Wgh   = (const float*)d_in[9];
  const float* bg    = (const float*)d_in[10];
  float* out = (float*)d_out;

  // Carve workspace (256B aligned slabs).
  char* p = (char*)d_ws;
  auto take = [&](size_t bytes) {
    char* r = p;
    p += (bytes + 255) & ~(size_t)255;
    return r;
  };
  unsigned*       bar    = (unsigned*)take(256);
  unsigned short* xb     = (unsigned short*)take((size_t)MM * DD * 2);
  unsigned short* Wxb    = (unsigned short*)take((size_t)DD * DD * 2);
  unsigned short* Rb     = (unsigned short*)take((size_t)DD * DD * 2);
  unsigned short* Wdb    = (unsigned short*)take((size_t)DD * DD * 2);
  unsigned short* Wgxb   = (unsigned short*)take((size_t)DD * DD * 2);
  unsigned short* Wghb   = (unsigned short*)take((size_t)DD * DD * 2);
  float*          decay  = (float*)take(DD * 4);
  float*          wxf    = (float*)take((size_t)MM * DD * 4);
  float*          alphaf = (float*)take((size_t)MM * DD * 4);
  unsigned short* gxb    = (unsigned short*)take((size_t)MM * DD * 2);
  unsigned short* hbt    = (unsigned short*)take((size_t)MM * DD * 2);
  unsigned short* hbuf   = (unsigned short*)take((size_t)2 * BB * DD * 2);

  // Precision conversions + derived constants.
  init_kernel<<<(BB * DD + 255) / 256, 256, 0, stream>>>(h0, hbuf, bar);
  cvt_f32_bf16_kernel<<<4096, 256, 0, stream>>>(x, xb, (size_t)MM * DD);
  cvt_f32_bf16_kernel<<<512, 256, 0, stream>>>(Wx, Wxb, (size_t)DD * DD);
  cvt_f32_bf16_kernel<<<512, 256, 0, stream>>>(R, Rb, (size_t)DD * DD);
  cvt_f32_bf16_kernel<<<512, 256, 0, stream>>>(Wd, Wdb, (size_t)DD * DD);
  cvt_f32_bf16_kernel<<<512, 256, 0, stream>>>(Wgx, Wgxb, (size_t)DD * DD);
  cvt_f32_bf16_kernel<<<512, 256, 0, stream>>>(Wgh, Wghb, (size_t)DD * DD);
  decay_kernel<<<4, 256, 0, stream>>>(Avec, decay);

  const int ggrid = (MM / GMT) * (DD / GNT);  // 2048 WGs
  // wx = x@Wx^T + bias
  gemm_epi_kernel<<<ggrid, 256, 0, stream>>>(xb, Wxb, bias, nullptr, nullptr,
                                             nullptr, wxf, nullptr, 0);
  // alpha = exp(-softplus(x@Wd^T + b_delta) * decay)
  gemm_epi_kernel<<<ggrid, 256, 0, stream>>>(xb, Wdb, bd, decay, nullptr,
                                             nullptr, alphaf, nullptr, 1);
  // gx = bf16(x@Wgx^T + b_gate)
  gemm_epi_kernel<<<ggrid, 256, 0, stream>>>(xb, Wgxb, bg, nullptr, nullptr,
                                             nullptr, nullptr, gxb, 2);
  // sequential recurrence (persistent, 16 WGs, 164KB LDS each, TDM staging)
  scan_kernel<<<SCAN_WGS, 256, SCAN_SMEM, stream>>>(Rb, wxf, alphaf, hbt, hbuf,
                                                    bar);
  // out = h * silu(gx + h@Wgh^T)
  gemm_epi_kernel<<<ggrid, 256, 0, stream>>>(hbt, Wghb, nullptr, nullptr, gxb,
                                             hbt, out, nullptr, 3);
}
